// Up_63917703299773
// MI455X (gfx1250) — matrix operand order repaired and checked by reading, not compiled, not run on hardware
//
#include <hip/hip_runtime.h>
#include <stdint.h>

// ---------------------------------------------------------------------------
// MI455X (gfx1250).  All convs (incl. deformable + offset predictors) are
// implicit GEMMs on v_wmma_f32_16x16x32_f16, 2x2 macro-tile per wave.
// Tap-major weights (k' = r*Cin + ci) make every 32-wide K step a constant
// kernel tap: B fragments are contiguous NHWC channel runs (b128 loads) and
// deformable bilinear state is computed once per tap.  Weight tiles staged by
// the Tensor Data Mover, double-buffered; loop peeled so the steady state is
// one tensor_load_to_lds + one s_wait_tensorcnt(1) + 4 ds b128 + 4 WMMA.
// ---------------------------------------------------------------------------

typedef __attribute__((ext_vector_type(16))) _Float16 v16h;
typedef __attribute__((ext_vector_type(8)))  _Float16 v8h;
typedef __attribute__((ext_vector_type(8)))  float    v8f;
typedef __attribute__((ext_vector_type(4)))  float    f32x4;
typedef __attribute__((ext_vector_type(4)))  unsigned int u32x4;
typedef __attribute__((ext_vector_type(8)))  int      i32x8;
typedef __attribute__((ext_vector_type(4)))  int      i32x4;

#if __has_builtin(__builtin_amdgcn_tensor_load_to_lds) && __has_builtin(__builtin_amdgcn_s_wait_tensorcnt)
#define HAVE_TDM 1
#else
#define HAVE_TDM 0
#endif

#define NB 4
#define HWPIX 4096
#define NPIX (NB*HWPIX)

__device__ __forceinline__ int a_kidx(int lane, int i) {
  return i + ((lane >> 4) << 4);
}

#if HAVE_TDM
__device__ __forceinline__ void tdm_load_32x32_f16(const _Float16* gsrc,
                                                   uint32_t lds_off,
                                                   uint32_t stride_elems) {
  uint64_t ga = (uint64_t)(uintptr_t)gsrc;
  u32x4 g0;
  g0[0] = 1u;
  g0[1] = lds_off;
  g0[2] = (uint32_t)ga;
  g0[3] = (uint32_t)((ga >> 32) & 0x01ffffffu) | (2u << 30);
  i32x8 g1;
  g1[0] = 0x00010000;                                    // data_size = 2B
  g1[1] = (int)((stride_elems & 0xffffu) << 16);         // tensor_dim0 lo
  g1[2] = (int)(((stride_elems >> 16) & 0xffffu) | (32u << 16)); // | tensor_dim1=32
  g1[3] = (int)(32u << 16);                              // tile_dim0 = 32
  g1[4] = 32;                                            // tile_dim1 = 32
  g1[5] = (int)stride_elems;                             // dim0 stride
  g1[6] = 0;
  g1[7] = 0;
  i32x4 z4 = {0, 0, 0, 0};
#if defined(__clang_major__) && (__clang_major__ >= 23)
  i32x8 z8 = {0, 0, 0, 0, 0, 0, 0, 0};
  __builtin_amdgcn_tensor_load_to_lds(g0, g1, z4, z4, z8, 0);
#else
  __builtin_amdgcn_tensor_load_to_lds(g0, g1, z4, z4, 0);
#endif
}
#endif

// Load one 16x32 A-fragment row pair from the LDS tile: exactly 2 b128 loads.
__device__ __forceinline__ v16h a_frag_lds(const _Float16* tile, int lane, int row16) {
  const v8h* ap = (const v8h*)(tile + (size_t)((lane & 15) + row16) * 32 + ((lane >> 4) << 4));
  v8h lo = ap[0], hi = ap[1];
  v16h a;
#pragma unroll
  for (int i = 0; i < 8; ++i) { a[i] = lo[i]; a[i + 8] = hi[i]; }
  return a;
}

// ---------------------------------------------------------------------------
// Weight permute + pad: dst[m][r*Cin+ci] = src[m][ci*KK+r]; rows >= CoutV -> 0
__global__ void k_w_perm(const float* __restrict__ src, _Float16* __restrict__ dst,
                         int CoutV, int CoutP, int Cin, int KK) {
  long long idx = (long long)blockIdx.x * 256 + threadIdx.x;
  long long n = (long long)CoutP * Cin * KK;
  if (idx >= n) return;
  int ck = Cin * KK;
  int m = (int)(idx / ck);
  int t = (int)(idx - (long long)m * ck);
  int r = t / Cin;
  int ci = t - r * Cin;
  dst[idx] = (m < CoutV) ? (_Float16)src[((size_t)m * Cin + ci) * KK + r]
                         : (_Float16)0.f;
}

// NCHW f32 -> NHWC f16 / f32 (C = 128)
__global__ void k_nhwc_f16(const float* __restrict__ src, _Float16* __restrict__ dst) {
  int idx = blockIdx.x * 256 + threadIdx.x;
  if (idx >= NB * 128 * HWPIX) return;
  int c = idx & 127, t = idx >> 7, hw = t & (HWPIX - 1), b = t >> 12;
  dst[idx] = (_Float16)src[(((size_t)b * 128 + c) << 12) + hw];
}
__global__ void k_nhwc_f32(const float* __restrict__ src, float* __restrict__ dst) {
  int idx = blockIdx.x * 256 + threadIdx.x;
  if (idx >= NB * 128 * HWPIX) return;
  int c = idx & 127, t = idx >> 7, hw = t & (HWPIX - 1), b = t >> 12;
  dst[idx] = src[(((size_t)b * 128 + c) << 12) + hw];
}

// Bilinear 2x upsample of x1 -> f16 NHWC channels [128,256) of dc1 input.
__global__ void k_upsample_to_cat(const float* __restrict__ x1,
                                  _Float16* __restrict__ cat2) {
  int idx = blockIdx.x * 256 + threadIdx.x;
  if (idx >= NB * 128 * HWPIX) return;
  int c = idx & 127, t = idx >> 7, hw = t & (HWPIX - 1), b = t >> 12;
  int oh = hw >> 6, ow = hw & 63;
  float py = oh * 31.0f / 63.0f, px = ow * 31.0f / 63.0f;
  int y0 = (int)floorf(py), x0 = (int)floorf(px);
  int y1 = y0 + 1 > 31 ? 31 : y0 + 1;
  int x1i = x0 + 1 > 31 ? 31 : x0 + 1;
  float wy = py - (float)y0, wx = px - (float)x0;
  const float* s = x1 + ((size_t)(b * 128 + c) << 10);
  float v = (1.f - wy) * (1.f - wx) * s[y0 * 32 + x0] +
            (1.f - wy) * wx * s[y0 * 32 + x1i] +
            wy * (1.f - wx) * s[y1 * 32 + x0] +
            wy * wx * s[y1 * 32 + x1i];
  cat2[(size_t)t * 256 + 128 + c] = (_Float16)v;
}

// ---------------------------------------------------------------------------
// Implicit-GEMM conv, NHWC activations, tap-major weights.
// grid = (NPIX/32, CoutPad/32), block = 32 (one wave).
template <int KH, int KW>
__global__ void __launch_bounds__(32)
k_conv_gemm(const _Float16* __restrict__ X, const _Float16* __restrict__ Wt,
            int Cin, int padH, int padW,
            const float* __restrict__ bng, const float* __restrict__ bnb,
            const float* __restrict__ bnm, const float* __restrict__ bnv,
            const float* __restrict__ bias, int relu,
            _Float16* __restrict__ outH, float* __restrict__ outF, int f32nchw,
            int coutOff, int CoutTot, int coutLim) {
  constexpr int KK = KH * KW;
  __shared__ _Float16 ldsA[2][1024];
  int lane = threadIdx.x;
  int pt = blockIdx.x, ct = blockIdx.y;
  int Ktot = Cin * KK;
  int steps = Ktot >> 5;                       // >= 4 always
  const _Float16* wrow = Wt + (size_t)ct * 32 * Ktot;
  int lhi = (lane >> 4) << 4;                  // 0 or 16 (channel sub-offset)

  int p0 = pt * 32 + (lane & 15), p1 = p0 + 16;
  int b0 = p0 >> 12, hw0 = p0 & (HWPIX - 1), h0 = hw0 >> 6, w0 = hw0 & 63;
  int b1 = p1 >> 12, hw1 = p1 & (HWPIX - 1), h1 = hw1 >> 6, w1 = hw1 & 63;

#if HAVE_TDM
  uint32_t lbase = (uint32_t)(uintptr_t)(&ldsA[0][0]);
  tdm_load_32x32_f16(wrow, lbase, (uint32_t)Ktot);
#endif
  int cur = 0;
  int crem = 0, ky = 0, kx = 0;                // tap-major walk over K
  v8f acc00 = {}, acc01 = {}, acc10 = {}, acc11 = {};

  auto body = [&](int s, int curb) {
    v16h a0, a1;
#if HAVE_TDM
    a0 = a_frag_lds(ldsA[curb], lane, 0);
    a1 = a_frag_lds(ldsA[curb], lane, 16);
#else
#pragma unroll
    for (int i = 0; i < 16; ++i) {
      int kk = a_kidx(lane, i);
      a0[i] = wrow[(size_t)(lane & 15) * Ktot + (s << 5) + kk];
      a1[i] = wrow[(size_t)((lane & 15) + 16) * Ktot + (s << 5) + kk];
    }
#endif
    // --- B: constant tap (ky,kx) this step; 16 consecutive NHWC channels ---
    int ih0 = h0 - padH + ky, iw0 = w0 - padW + kx;
    int ih1 = h1 - padH + ky, iw1 = w1 - padW + kx;
    bool ok0 = ((unsigned)ih0 < 64u) & ((unsigned)iw0 < 64u);
    bool ok1 = ((unsigned)ih1 < 64u) & ((unsigned)iw1 < 64u);
    v16h bf0, bf1;
    if (ok0) {
      const v8h* sp = (const v8h*)(X + (size_t)(b0 * HWPIX + ih0 * 64 + iw0) * Cin + crem + lhi);
      v8h lo = sp[0], hi = sp[1];
#pragma unroll
      for (int i = 0; i < 8; ++i) { bf0[i] = lo[i]; bf0[i + 8] = hi[i]; }
    } else {
#pragma unroll
      for (int i = 0; i < 16; ++i) bf0[i] = (_Float16)0.f;
    }
    if (ok1) {
      const v8h* sp = (const v8h*)(X + (size_t)(b1 * HWPIX + ih1 * 64 + iw1) * Cin + crem + lhi);
      v8h lo = sp[0], hi = sp[1];
#pragma unroll
      for (int i = 0; i < 8; ++i) { bf1[i] = lo[i]; bf1[i + 8] = hi[i]; }
    } else {
#pragma unroll
      for (int i = 0; i < 16; ++i) bf1[i] = (_Float16)0.f;
    }
    acc00 = __builtin_amdgcn_wmma_f32_16x16x32_f16(false, a0, false, bf0, (short)0, acc00, false, false);
    acc10 = __builtin_amdgcn_wmma_f32_16x16x32_f16(false, a1, false, bf0, (short)0, acc10, false, false);
    acc01 = __builtin_amdgcn_wmma_f32_16x16x32_f16(false, a0, false, bf1, (short)0, acc01, false, false);
    acc11 = __builtin_amdgcn_wmma_f32_16x16x32_f16(false, a1, false, bf1, (short)0, acc11, false, false);
    crem += 32;
    if (crem >= Cin) {
      crem = 0;
      if (++kx >= KW) { kx = 0; ++ky; }
    }
  };

#if HAVE_TDM
  for (int s = 0; s < steps - 1; ++s) {        // steady state: 1 TDM + 1 wait
    tdm_load_32x32_f16(wrow + (size_t)((s + 1) << 5),
                       lbase + (uint32_t)((cur ^ 1) * 2048), (uint32_t)Ktot);
    __builtin_amdgcn_s_wait_tensorcnt(1);
    body(s, cur);
    cur ^= 1;
  }
  __builtin_amdgcn_s_wait_tensorcnt(0);        // peeled last step
  body(steps - 1, cur);
#else
  for (int s = 0; s < steps; ++s) body(s, 0);
#endif

  // epilogue: 8 consecutive Cout channels per lane -> one b128 store (NHWC)
  auto emit = [&](const v8f& a, int mbase, int b, int hw) {
    int cb = mbase + ((lane >> 4) << 3);
    if (outH) {
      v8h ov;
#pragma unroll
      for (int rr = 0; rr < 8; ++rr) {
        int c = cb + rr;
        float y = a[rr];
        if (bng) y = bng[c] * (y - bnm[c]) * rsqrtf(bnv[c] + 1e-5f) + bnb[c];
        if (relu) y = fmaxf(y, 0.f);
        ov[rr] = (_Float16)y;
      }
      *(v8h*)(outH + (size_t)(b * HWPIX + hw) * CoutTot + coutOff + cb) = ov;
    } else {
#pragma unroll
      for (int rr = 0; rr < 8; ++rr) {
        int c = cb + rr;
        if (c >= coutLim) continue;
        float y = a[rr];
        if (bias) y += bias[c];
        if (bng) y = bng[c] * (y - bnm[c]) * rsqrtf(bnv[c] + 1e-5f) + bnb[c];
        if (relu) y = fmaxf(y, 0.f);
        if (f32nchw) outF[(((size_t)b * CoutTot + coutOff + c) << 12) + hw] = y;
        else         outF[(size_t)(b * HWPIX + hw) * CoutTot + coutOff + c] = y;
      }
    }
  };
  emit(acc00, ct * 32,      b0, hw0);
  emit(acc10, ct * 32 + 16, b0, hw0);
  emit(acc01, ct * 32,      b1, hw1);
  emit(acc11, ct * 32 + 16, b1, hw1);
}

// ---------------------------------------------------------------------------
// Deformable conv (Cin=Cout=128) GEMM.  Bilinear state computed once per tap
// (every 4 K-steps); per element: 4 contiguous-f32 corner reads + 4 FMAs.
struct Samp { int o[4]; float w[4]; };
__device__ __forceinline__ Samp mk_samp(const float* __restrict__ OFF, int twoK,
                                        int b, int hw, int h, int w,
                                        int padH, int padW, int ky, int kx, int r) {
  float dy = OFF[(size_t)(b * HWPIX + hw) * twoK + 2 * r];
  float dx = OFF[(size_t)(b * HWPIX + hw) * twoK + 2 * r + 1];
  float py = (float)(h - padH + ky) + dy;
  float px = (float)(w - padW + kx) + dx;
  float y0f = floorf(py), x0f = floorf(px);
  float ay = py - y0f, ax = px - x0f;
  int y0 = (int)y0f, x0 = (int)x0f;
  int ys[2] = {y0, y0 + 1};
  int xs[2] = {x0, x0 + 1};
  float wy[2] = {1.f - ay, ay};
  float wx[2] = {1.f - ax, ax};
  Samp sm;
#pragma unroll
  for (int jy = 0; jy < 2; ++jy)
#pragma unroll
    for (int jx = 0; jx < 2; ++jx) {
      int yy = ys[jy], xx = xs[jx];
      bool v = (yy >= 0) && (yy <= 63) && (xx >= 0) && (xx <= 63);
      int yc = yy < 0 ? 0 : (yy > 63 ? 63 : yy);
      int xc = xx < 0 ? 0 : (xx > 63 ? 63 : xx);
      sm.o[jy * 2 + jx] = (b * HWPIX + yc * 64 + xc) * 128;
      sm.w[jy * 2 + jx] = v ? wy[jy] * wx[jx] : 0.f;
    }
  return sm;
}

__device__ __forceinline__ void dcn_bfrag(v16h& bf, const float* __restrict__ Xn,
                                          const Samp& sm, int cib) {
  f32x4 c0[4], c1[4], c2[4], c3[4];
#pragma unroll
  for (int j = 0; j < 4; ++j) {
    c0[j] = *(const f32x4*)(Xn + sm.o[0] + cib + 4 * j);
    c1[j] = *(const f32x4*)(Xn + sm.o[1] + cib + 4 * j);
    c2[j] = *(const f32x4*)(Xn + sm.o[2] + cib + 4 * j);
    c3[j] = *(const f32x4*)(Xn + sm.o[3] + cib + 4 * j);
  }
#pragma unroll
  for (int i = 0; i < 16; ++i) {
    float v = c0[i >> 2][i & 3] * sm.w[0] + c1[i >> 2][i & 3] * sm.w[1] +
              c2[i >> 2][i & 3] * sm.w[2] + c3[i >> 2][i & 3] * sm.w[3];
    bf[i] = (_Float16)v;
  }
}

template <int K>
__global__ void __launch_bounds__(32)
k_dcn_gemm(const float* __restrict__ Xn, const float* __restrict__ OFF,
           const _Float16* __restrict__ Wt, int KW, int padH, int padW,
           const float* __restrict__ bng, const float* __restrict__ bnb,
           const float* __restrict__ bnm, const float* __restrict__ bnv,
           _Float16* __restrict__ outH, int coutOff, int CoutTot) {
  __shared__ _Float16 ldsA[2][1024];
  int lane = threadIdx.x;
  int pt = blockIdx.x, ct = blockIdx.y;
  constexpr int Ktot = 128 * K;
  constexpr int steps = Ktot >> 5;
  const _Float16* wrow = Wt + (size_t)ct * 32 * Ktot;
  int lhi = (lane >> 4) << 4;

  int p0 = pt * 32 + (lane & 15), p1 = p0 + 16;
  int b0 = p0 >> 12, hw0 = p0 & (HWPIX - 1), h0 = hw0 >> 6, w0 = hw0 & 63;
  int b1 = p1 >> 12, hw1 = p1 & (HWPIX - 1), h1 = hw1 >> 6, w1 = hw1 & 63;

#if HAVE_TDM
  uint32_t lbase = (uint32_t)(uintptr_t)(&ldsA[0][0]);
  tdm_load_32x32_f16(wrow, lbase, (uint32_t)Ktot);
#endif
  int cur = 0;
  int ky = 0, kx = 0, r = 0;
  Samp sp0 = {}, sp1 = {};
  v8f acc00 = {}, acc01 = {}, acc10 = {}, acc11 = {};

  auto body = [&](int s, int curb) {
    v16h a0, a1;
#if HAVE_TDM
    a0 = a_frag_lds(ldsA[curb], lane, 0);
    a1 = a_frag_lds(ldsA[curb], lane, 16);
#else
#pragma unroll
    for (int i = 0; i < 16; ++i) {
      int kk = a_kidx(lane, i);
      a0[i] = wrow[(size_t)(lane & 15) * Ktot + (s << 5) + kk];
      a1[i] = wrow[(size_t)((lane & 15) + 16) * Ktot + (s << 5) + kk];
    }
#endif
    if ((s & 3) == 0) {          // new tap: recompute bilinear sampling state
      sp0 = mk_samp(OFF, 2 * K, b0, hw0, h0, w0, padH, padW, ky, kx, r);
      sp1 = mk_samp(OFF, 2 * K, b1, hw1, h1, w1, padH, padW, ky, kx, r);
    }
    int cib = ((s & 3) << 5) + lhi;
    v16h bf0, bf1;
    dcn_bfrag(bf0, Xn, sp0, cib);
    dcn_bfrag(bf1, Xn, sp1, cib);
    acc00 = __builtin_amdgcn_wmma_f32_16x16x32_f16(false, a0, false, bf0, (short)0, acc00, false, false);
    acc10 = __builtin_amdgcn_wmma_f32_16x16x32_f16(false, a1, false, bf0, (short)0, acc10, false, false);
    acc01 = __builtin_amdgcn_wmma_f32_16x16x32_f16(false, a0, false, bf1, (short)0, acc01, false, false);
    acc11 = __builtin_amdgcn_wmma_f32_16x16x32_f16(false, a1, false, bf1, (short)0, acc11, false, false);
    if ((s & 3) == 3) {
      ++r;
      if (++kx >= KW) { kx = 0; ++ky; }
    }
  };

#if HAVE_TDM
  for (int s = 0; s < steps - 1; ++s) {
    tdm_load_32x32_f16(wrow + (size_t)((s + 1) << 5),
                       lbase + (uint32_t)((cur ^ 1) * 2048), (uint32_t)Ktot);
    __builtin_amdgcn_s_wait_tensorcnt(1);
    body(s, cur);
    cur ^= 1;
  }
  __builtin_amdgcn_s_wait_tensorcnt(0);
  body(steps - 1, cur);
#else
  for (int s = 0; s < steps; ++s) body(s, 0);
#endif

  auto emit = [&](const v8f& a, int mbase, int b, int hw) {
    int cb = mbase + ((lane >> 4) << 3);
    v8h ov;
#pragma unroll
    for (int rr = 0; rr < 8; ++rr) {
      int c = cb + rr;
      float y = a[rr];
      y = bng[c] * (y - bnm[c]) * rsqrtf(bnv[c] + 1e-5f) + bnb[c];
      y = fmaxf(y, 0.f);
      ov[rr] = (_Float16)y;
    }
    *(v8h*)(outH + (size_t)(b * HWPIX + hw) * CoutTot + coutOff + cb) = ov;
  };
  emit(acc00, ct * 32,      b0, hw0);
  emit(acc10, ct * 32 + 16, b0, hw0);
  emit(acc01, ct * 32,      b1, hw1);
  emit(acc11, ct * 32 + 16, b1, hw1);
}

// ---------------------------------------------------------------------------
extern "C" void kernel_launch(void* const* d_in, const int* in_sizes, int n_in,
                              void* d_out, int out_size, void* d_ws, size_t ws_size,
                              hipStream_t stream) {
  (void)in_sizes; (void)n_in; (void)out_size; (void)ws_size;
  auto F = [&](int i) { return (const float*)d_in[i]; };

  char* ws = (char*)d_ws;
  size_t cursor = 0;
  auto alloc = [&](size_t bytes) -> char* {
    size_t o = (cursor + 255) & ~(size_t)255;
    cursor = o + bytes;
    return ws + o;
  };

  _Float16* x2h   = (_Float16*)alloc((size_t)NB * HWPIX * 128 * 2);   // NHWC f16
  float*    x2n   = (float*)alloc((size_t)NB * HWPIX * 128 * 4);      // NHWC f32
  _Float16* cat10 = (_Float16*)alloc((size_t)NB * HWPIX * 1280 * 2);  // NHWC
  _Float16* cat2  = (_Float16*)alloc((size_t)NB * HWPIX * 256 * 2);   // NHWC
  _Float16* dc1o  = (_Float16*)alloc((size_t)NB * HWPIX * 128 * 2);   // NHWC
  float*    offs  = (float*)alloc((size_t)NB * HWPIX * 70 * 4);       // NHWC

  auto wperm = [&](int idx, int CoutV, int CoutP, int Cin, int KK) -> _Float16* {
    size_t n = (size_t)CoutP * Cin * KK;
    _Float16* dst = (_Float16*)alloc(n * 2);
    k_w_perm<<<(unsigned)((n + 255) / 256), 256, 0, stream>>>(F(idx), dst, CoutV, CoutP, Cin, KK);
    return dst;
  };
  _Float16* w_dc1   = wperm(6, 128, 128, 256, 9);
  _Float16* w_dc2   = wperm(11, 128, 128, 128, 9);
  _Float16* w_fus   = wperm(62, 128, 128, 1280, 1);
  _Float16* w_de1   = wperm(12, 128, 128, 128, 1);
  _Float16* w_de3eb = wperm(13, 128, 128, 128, 1);
  _Float16* w_de3eh = wperm(18, 128, 128, 128, 3);
  _Float16* w_de3ev = wperm(25, 128, 128, 128, 3);
  _Float16* w_de5eb = wperm(28, 128, 128, 128, 9);
  _Float16* w_de5eh = wperm(33, 128, 128, 128, 15);
  _Float16* w_de5ev = wperm(40, 128, 128, 128, 15);
  _Float16* w_de7eb = wperm(43, 128, 128, 128, 25);
  _Float16* w_de7eh = wperm(48, 128, 128, 128, 35);
  _Float16* w_de7ev = wperm(55, 128, 128, 128, 35);
  _Float16* w_o3h = wperm(20, 6, 32, 128, 9);
  _Float16* w_o3v = wperm(27, 6, 32, 128, 9);
  _Float16* w_o5h = wperm(35, 30, 32, 128, 9);
  _Float16* w_o5v = wperm(42, 30, 32, 128, 9);
  _Float16* w_o7h = wperm(50, 70, 96, 128, 9);
  _Float16* w_o7v = wperm(57, 70, 96, 128, 9);

  const int nx = NB * 128 * HWPIX;
  k_nhwc_f16<<<(nx + 255) / 256, 256, 0, stream>>>(F(1), x2h);
  k_nhwc_f32<<<(nx + 255) / 256, 256, 0, stream>>>(F(1), x2n);
  k_upsample_to_cat<<<(nx + 255) / 256, 256, 0, stream>>>(F(0), cat2);

  dim3 gg(NPIX / 32, 4), gb(32);
  auto off_gemm = [&](_Float16* ow, int bidx, int Co, int padRows) {
    dim3 g(NPIX / 32, padRows / 32);
    k_conv_gemm<3, 3><<<g, gb, 0, stream>>>(x2h, ow, 128, 1, 1,
        nullptr, nullptr, nullptr, nullptr, F(bidx), 0,
        nullptr, offs, 0, 0, Co, Co);
  };

  // ---- MSDE -> 1280-ch NHWC concat ----
  k_conv_gemm<1, 1><<<gg, gb, 0, stream>>>(x2h, w_de1, 128, 0, 0,
      nullptr, nullptr, nullptr, nullptr, nullptr, 0, cat10, nullptr, 0, 0, 1280, 128);
  k_conv_gemm<1, 1><<<gg, gb, 0, stream>>>(x2h, w_de3eb, 128, 0, 0,
      nullptr, nullptr, nullptr, nullptr, nullptr, 0, cat10, nullptr, 0, 128, 1280, 128);
  off_gemm(w_o3h, 19, 6, 32);
  k_dcn_gemm<3><<<gg, gb, 0, stream>>>(x2n, offs, w_de3eh, 3, 0, 1,
      F(14), F(15), F(16), F(17), cat10, 256, 1280);
  off_gemm(w_o3v, 26, 6, 32);
  k_dcn_gemm<3><<<gg, gb, 0, stream>>>(x2n, offs, w_de3ev, 1, 1, 0,
      F(21), F(22), F(23), F(24), cat10, 384, 1280);
  k_conv_gemm<3, 3><<<gg, gb, 0, stream>>>(x2h, w_de5eb, 128, 1, 1,
      nullptr, nullptr, nullptr, nullptr, nullptr, 0, cat10, nullptr, 0, 512, 1280, 128);
  off_gemm(w_o5h, 34, 30, 32);
  k_dcn_gemm<15><<<gg, gb, 0, stream>>>(x2n, offs, w_de5eh, 5, 1, 2,
      F(29), F(30), F(31), F(32), cat10, 640, 1280);
  off_gemm(w_o5v, 41, 30, 32);
  k_dcn_gemm<15><<<gg, gb, 0, stream>>>(x2n, offs, w_de5ev, 3, 2, 1,
      F(36), F(37), F(38), F(39), cat10, 768, 1280);
  k_conv_gemm<5, 5><<<gg, gb, 0, stream>>>(x2h, w_de7eb, 128, 2, 2,
      nullptr, nullptr, nullptr, nullptr, nullptr, 0, cat10, nullptr, 0, 896, 1280, 128);
  off_gemm(w_o7h, 49, 70, 96);
  k_dcn_gemm<35><<<gg, gb, 0, stream>>>(x2n, offs, w_de7eh, 7, 2, 3,
      F(44), F(45), F(46), F(47), cat10, 1024, 1280);
  off_gemm(w_o7v, 56, 70, 96);
  k_dcn_gemm<35><<<gg, gb, 0, stream>>>(x2n, offs, w_de7ev, 5, 3, 2,
      F(51), F(52), F(53), F(54), cat10, 1152, 1280);

  // fus 1x1 over 1280 ch, BN+ReLU -> NHWC channels [0,128) of dc1 input
  k_conv_gemm<1, 1><<<gg, gb, 0, stream>>>(cat10, w_fus, 1280, 0, 0,
      F(58), F(59), F(60), F(61), nullptr, 1, cat2, nullptr, 0, 0, 256, 128);
  // dc1 3x3 pad1 over 256 ch, BN+ReLU -> NHWC f16
  k_conv_gemm<3, 3><<<gg, gb, 0, stream>>>(cat2, w_dc1, 256, 1, 1,
      F(2), F(3), F(4), F(5), nullptr, 1, dc1o, nullptr, 0, 0, 128, 128);
  // dc2 3x3 pad1 over 128 ch, BN+ReLU -> NCHW fp32 final output
  k_conv_gemm<3, 3><<<gg, gb, 0, stream>>>(dc1o, w_dc2, 128, 1, 1,
      F(7), F(8), F(9), F(10), nullptr, 1, nullptr, (float*)d_out, 1, 0, 128, 128);
}
